// SentenceFilterer_9509057593544
// MI455X (gfx1250) — matrix operand order repaired and checked by reading, not compiled, not run on hardware
//
#include <hip/hip_runtime.h>
#include <hip/hip_bf16.h>
#include <stdint.h>

// Problem constants (match reference)
#define HH   512      // hidden
#define D3   1536     // 3*H
#define DD   1024     // input dim = 2*H
#define BB   8
#define NSEC_ 4
#define SS   32
#define LL   64
#define NSENT 1024    // B*NSEC*S
#define TDOC  128     // NSEC*S
#define SECSZ 2048    // S*L

typedef __attribute__((ext_vector_type(16))) __bf16 v16bf;
typedef __attribute__((ext_vector_type(8)))  float  v8f;

union FragBF { v16bf v; uint4 q[2]; };
union AccF   { v8f v; float f[8]; };

__device__ __forceinline__ unsigned short f2bf(float x) {
  union { float f; unsigned int u; } c; c.f = x;
  unsigned int u = c.u;
  return (unsigned short)((u + 0x7FFFu + ((u >> 16) & 1u)) >> 16);
}
__device__ __forceinline__ float sigmoidf_(float x) { return 1.0f / (1.0f + __expf(-x)); }

__device__ __forceinline__ v8f wmma_bf16(v8f acc, const FragBF& A, const FragBF& Bm) {
  return __builtin_amdgcn_wmma_f32_16x16x32_bf16(false, A.v, false, Bm.v, (short)0, acc, false, false);
}

// ---------------- f32 -> bf16 convert (8 elems / thread) ----------------
__global__ void cvt_bf16_kernel(const float* __restrict__ in, unsigned short* __restrict__ out, int n8) {
  int i = blockIdx.x * blockDim.x + threadIdx.x;
  if (i >= n8) return;
  const float4* p = (const float4*)in + (size_t)i * 2;
  float4 a = p[0], b = p[1];
  union { unsigned short s[8]; uint4 q; } u;
  u.s[0] = f2bf(a.x); u.s[1] = f2bf(a.y); u.s[2] = f2bf(a.z); u.s[3] = f2bf(a.w);
  u.s[4] = f2bf(b.x); u.s[5] = f2bf(b.y); u.s[6] = f2bf(b.z); u.s[7] = f2bf(b.w);
  ((uint4*)out)[i] = u.q;
}

// ---------------- bias prep: comb (r,z: bih+bhh; n: bih), hn = bhh_n ----
__global__ void bias_prep_kernel(const float* __restrict__ bih, const float* __restrict__ bhh,
                                 float* __restrict__ comb, float* __restrict__ hn) {
  int g = blockIdx.x * blockDim.x + threadIdx.x;
  if (g >= D3) return;
  float a = bih[g], b = bhh[g];
  comb[g] = (g < 2 * HH) ? (a + b) : a;
  if (g >= 2 * HH) hn[g - 2 * HH] = b;
}

// LDS strides (16B-aligned rows, bank-staggered)
#define SA_STRIDE 1040   // ushort
#define SH_STRIDE 528    // ushort
#define SG_STRIDE 1552   // float
#define SN_STRIDE 520    // float
#define SENT_SMEM (16*SA_STRIDE*2 + 16*SH_STRIDE*2 + 16*SG_STRIDE*4 + 16*SN_STRIDE*4 + 16*SN_STRIDE*4)
#define DOC_SMEM  (16*SG_STRIDE*4 + 16*SN_STRIDE*4 + 16*SN_STRIDE*4 + 16*SH_STRIDE*2)

// ---------------- sentence-level GRU (fused input proj + recurrence) ----
// grid: 128 blocks = dir(2) x rowTile(64); block: 256 threads (8 waves)
// Per wave: 12 gate-column tiles, 12 live f32 accumulators; A frag shared
// across all 12 tiles per K-step; B frags pre-loaded in bursts of 6 so the
// 6 WMMAs issue back-to-back on independent accumulators.
__global__ void __launch_bounds__(256, 1)
sent_gru_kernel(const unsigned short* __restrict__ xbf,   // [NSENT*64, 1024] bf16
                const unsigned short* __restrict__ Wih,   // [2][1536*1024] bf16
                const unsigned short* __restrict__ Whh,   // [2][1536*512]  bf16
                const float* __restrict__ biasComb,       // [2][1536]
                const float* __restrict__ biasHn,         // [2][512]
                unsigned short* __restrict__ sent_h)      // [1024][1024] bf16
{
  extern __shared__ char smem[];
  unsigned short* sA  = (unsigned short*)smem;
  unsigned short* sHb = sA + 16 * SA_STRIDE;
  float* sG  = (float*)(sHb + 16 * SH_STRIDE);
  float* sNh = sG + 16 * SG_STRIDE;
  float* sHf = sNh + 16 * SN_STRIDE;

  const int tid = threadIdx.x;
  const int wave = tid >> 5, lane = tid & 31;
  const int dir = blockIdx.x >> 6;
  const int n0 = (blockIdx.x & 63) * 16;

  const unsigned short* WihD = Wih + (size_t)dir * D3 * DD;
  const unsigned short* WhhD = Whh + (size_t)dir * D3 * HH;
  const float* bC = biasComb + dir * D3;
  const float* bN = biasHn + dir * HH;

  const int ln15 = lane & 15;
  const int kcA = (lane < 16) ? 0 : 8;
  const int kbB = (lane < 16) ? 0 : 8;
  const int mBase = (lane < 16) ? 0 : 8;
  const int gbase = wave * 192;                     // wave*12*16

  // hoist biases + weight base addresses out of the time loop
  float bx[12], bh[12];
  #pragma unroll
  for (int j = 0; j < 12; ++j) {
    int g0 = gbase + j * 16;
    bx[j] = bC[g0 + ln15];
    bh[j] = (g0 >= 2 * HH) ? bN[g0 - 2 * HH + ln15] : 0.0f;
  }
  const unsigned short* wihBase = WihD + (size_t)(gbase + ln15) * DD + kbB;
  const unsigned short* whhBase = WhhD + (size_t)(gbase + ln15) * HH + kbB;

  for (int e = tid; e < 16 * HH; e += 256) {
    int m = e >> 9, c = e & (HH - 1);
    sHf[m * SN_STRIDE + c] = 0.0f;
    sHb[m * SH_STRIDE + c] = 0;
  }
  __syncthreads();

  for (int step = 0; step < LL; ++step) {
    const int t = dir ? (LL - 1 - step) : step;
    // stage x_t tile [16 x 1024] bf16 into LDS (16B chunks)
    for (int i = 0; i < 8; ++i) {
      int e8 = i * 256 + tid;
      int m = e8 >> 7;
      int c = (e8 & 127) << 3;
      *(uint4*)(sA + m * SA_STRIDE + c) =
          *(const uint4*)(xbf + ((size_t)(n0 + m) * LL + t) * DD + c);
    }
    __syncthreads();

    AccF acc[12];
    #pragma unroll
    for (int j = 0; j < 12; ++j)
      #pragma unroll
      for (int r = 0; r < 8; ++r) acc[j].f[r] = bx[j];

    // ---- input projection: K = 1024 over Wih ----
    const unsigned short* arow = sA + ln15 * SA_STRIDE + kcA;
    #pragma unroll 2
    for (int k0 = 0; k0 < DD; k0 += 32) {
      FragBF A;
      A.q[0] = *(const uint4*)(arow + k0);
      A.q[1] = *(const uint4*)(arow + k0 + 16);
      const unsigned short* wp = wihBase + k0;
      __builtin_prefetch(wp + 2048, 0, 1);          // same k, two weight rows ahead
      #pragma unroll
      for (int jg = 0; jg < 2; ++jg) {
        FragBF Bf[6];
        #pragma unroll
        for (int j = 0; j < 6; ++j) {
          const unsigned short* w = wp + (size_t)(jg * 6 + j) * (16 * DD);
          Bf[j].q[0] = *(const uint4*)(w);
          Bf[j].q[1] = *(const uint4*)(w + 16);
        }
        #pragma unroll
        for (int j = 0; j < 6; ++j)
          acc[jg * 6 + j].v = wmma_bf16(acc[jg * 6 + j].v, A, Bf[j]);
      }
    }
    // stash xg for all tiles
    #pragma unroll
    for (int j = 0; j < 12; ++j) {
      int g0 = gbase + j * 16;
      #pragma unroll
      for (int r = 0; r < 8; ++r)
        sG[(mBase + r) * SG_STRIDE + g0 + ln15] = acc[j].f[r];
    }

    // ---- recurrence: K = 512 over Whh ----
    #pragma unroll
    for (int j = 0; j < 12; ++j)
      #pragma unroll
      for (int r = 0; r < 8; ++r) acc[j].f[r] = bh[j];

    const unsigned short* hrow = sHb + ln15 * SH_STRIDE + kcA;
    #pragma unroll 2
    for (int k0 = 0; k0 < HH; k0 += 32) {
      FragBF A;
      A.q[0] = *(const uint4*)(hrow + k0);
      A.q[1] = *(const uint4*)(hrow + k0 + 16);
      const unsigned short* vp = whhBase + k0;
      __builtin_prefetch(vp + 2048, 0, 1);
      #pragma unroll
      for (int jg = 0; jg < 2; ++jg) {
        FragBF Bf[6];
        #pragma unroll
        for (int j = 0; j < 6; ++j) {
          const unsigned short* w = vp + (size_t)(jg * 6 + j) * (16 * HH);
          Bf[j].q[0] = *(const uint4*)(w);
          Bf[j].q[1] = *(const uint4*)(w + 16);
        }
        #pragma unroll
        for (int j = 0; j < 6; ++j)
          acc[jg * 6 + j].v = wmma_bf16(acc[jg * 6 + j].v, A, Bf[j]);
      }
    }
    // combine: r/z tiles add hg into sG; n tiles keep hg separate in sNh
    #pragma unroll
    for (int j = 0; j < 12; ++j) {
      int g0 = gbase + j * 16;
      if (g0 < 2 * HH) {
        #pragma unroll
        for (int r = 0; r < 8; ++r)
          sG[(mBase + r) * SG_STRIDE + g0 + ln15] += acc[j].f[r];
      } else {
        #pragma unroll
        for (int r = 0; r < 8; ++r)
          sNh[(mBase + r) * SN_STRIDE + (g0 - 2 * HH) + ln15] = acc[j].f[r];
      }
    }
    __syncthreads();

    // gate update: h_new = (1-z)*tanh(xn + r*hn) + z*h
    for (int i = 0; i < 32; ++i) {
      int e = i * 256 + tid;
      int m = e >> 9, c = e & (HH - 1);
      float rr = sigmoidf_(sG[m * SG_STRIDE + c]);
      float zz = sigmoidf_(sG[m * SG_STRIDE + HH + c]);
      float nn = tanhf(sG[m * SG_STRIDE + 2 * HH + c] + rr * sNh[m * SN_STRIDE + c]);
      float hnew = (1.0f - zz) * nn + zz * sHf[m * SN_STRIDE + c];
      sHf[m * SN_STRIDE + c] = hnew;
      sHb[m * SH_STRIDE + c] = f2bf(hnew);
    }
    __syncthreads();
  }

  for (int e = tid; e < 16 * HH; e += 256) {
    int m = e >> 9, c = e & (HH - 1);
    sent_h[(size_t)(n0 + m) * DD + dir * HH + c] = sHb[m * SH_STRIDE + c];
  }
}

// ---------------- doc input projection: dg = sent_h @ Wih_d^T + bias ----
// grid (96, 64, 2), block 32 (one wave per 16x16 tile)
__global__ void doc_proj_kernel(const unsigned short* __restrict__ sent_h, // [1024][1024] bf16
                                const unsigned short* __restrict__ WihD2,  // [2][1536*1024] bf16
                                const float* __restrict__ biasComb,        // [2][1536]
                                float* __restrict__ dg)                     // [2][1024][1536]
{
  const int lane = threadIdx.x & 31;
  const int g0 = blockIdx.x * 16;
  const int row0 = blockIdx.y * 16;
  const int dir = blockIdx.z;
  const unsigned short* W = WihD2 + (size_t)dir * D3 * DD;
  const int ln15 = lane & 15;
  const int kcA = (lane < 16) ? 0 : 8;
  const int kbB = (lane < 16) ? 0 : 8;
  AccF acc;
  float bx = biasComb[dir * D3 + g0 + ln15];
  #pragma unroll
  for (int r = 0; r < 8; ++r) acc.f[r] = bx;
  const unsigned short* arow = sent_h + (size_t)(row0 + ln15) * DD + kcA;
  const unsigned short* wrow = W + (size_t)(g0 + ln15) * DD + kbB;
  #pragma unroll 4
  for (int k0 = 0; k0 < DD; k0 += 32) {
    FragBF A, Bm;
    A.q[0]  = *(const uint4*)(arow + k0);
    A.q[1]  = *(const uint4*)(arow + k0 + 16);
    Bm.q[0] = *(const uint4*)(wrow + k0);
    Bm.q[1] = *(const uint4*)(wrow + k0 + 16);
    acc.v = wmma_bf16(acc.v, A, Bm);
  }
  const int mBase = (lane < 16) ? 0 : 8;
  float* out = dg + (size_t)dir * NSENT * D3;
  #pragma unroll
  for (int r = 0; r < 8; ++r)
    out[(size_t)(row0 + mBase + r) * D3 + g0 + ln15] = acc.f[r];
}

// ---------------- doc-level GRU recurrence (xg precomputed in dg) -------
// grid 2 (dir), block 256 (8 waves); M=8 docs padded to 16
__global__ void __launch_bounds__(256, 1)
doc_gru_kernel(const float* __restrict__ dg,            // [2][1024][1536]
               const unsigned short* __restrict__ Whh,  // [2][1536*512] bf16
               const float* __restrict__ biasHn,        // [2][512]
               float* __restrict__ doc_out)             // [8][128][1024]
{
  extern __shared__ char smem[];
  float* sG  = (float*)smem;
  float* sNh = sG + 16 * SG_STRIDE;
  float* sHf = sNh + 16 * SN_STRIDE;
  unsigned short* sHb = (unsigned short*)(sHf + 16 * SN_STRIDE);

  const int tid = threadIdx.x, wave = tid >> 5, lane = tid & 31;
  const int dir = blockIdx.x;
  const float* dgD = dg + (size_t)dir * NSENT * D3;
  const unsigned short* WhhD = Whh + (size_t)dir * D3 * HH;
  const float* bN = biasHn + dir * HH;
  const int ln15 = lane & 15;
  const int kcA = (lane < 16) ? 0 : 8;
  const int kbB = (lane < 16) ? 0 : 8;
  const int gbase = wave * 192;

  float bh[12];
  #pragma unroll
  for (int j = 0; j < 12; ++j) {
    int g0 = gbase + j * 16;
    bh[j] = (g0 >= 2 * HH) ? bN[g0 - 2 * HH + ln15] : 0.0f;
  }
  const unsigned short* whhBase = WhhD + (size_t)(gbase + ln15) * HH + kbB;

  for (int e = tid; e < 16 * HH; e += 256) {
    int m = e >> 9, c = e & (HH - 1);
    sHf[m * SN_STRIDE + c] = 0.0f;
    sHb[m * SH_STRIDE + c] = 0;       // rows 8..15 stay zero (padding)
  }
  __syncthreads();

  for (int step = 0; step < TDOC; ++step) {
    const int t = dir ? (TDOC - 1 - step) : step;
    for (int m = 0; m < 8; ++m)
      for (int i = 0; i < 6; ++i) {
        int c = i * 256 + tid;
        sG[m * SG_STRIDE + c] = dgD[((size_t)(m * TDOC + t)) * D3 + c];
      }
    __syncthreads();

    AccF acc[12];
    #pragma unroll
    for (int j = 0; j < 12; ++j)
      #pragma unroll
      for (int r = 0; r < 8; ++r) acc[j].f[r] = bh[j];

    const unsigned short* hrow = sHb + ln15 * SH_STRIDE + kcA;
    #pragma unroll 2
    for (int k0 = 0; k0 < HH; k0 += 32) {
      FragBF A;
      A.q[0] = *(const uint4*)(hrow + k0);
      A.q[1] = *(const uint4*)(hrow + k0 + 16);
      const unsigned short* vp = whhBase + k0;
      #pragma unroll
      for (int jg = 0; jg < 2; ++jg) {
        FragBF Bf[6];
        #pragma unroll
        for (int j = 0; j < 6; ++j) {
          const unsigned short* w = vp + (size_t)(jg * 6 + j) * (16 * HH);
          Bf[j].q[0] = *(const uint4*)(w);
          Bf[j].q[1] = *(const uint4*)(w + 16);
        }
        #pragma unroll
        for (int j = 0; j < 6; ++j)
          acc[jg * 6 + j].v = wmma_bf16(acc[jg * 6 + j].v, A, Bf[j]);
      }
    }
    if (lane < 16) {                  // rows 0..7 valid (m = r)
      #pragma unroll
      for (int j = 0; j < 12; ++j) {
        int g0 = gbase + j * 16;
        if (g0 < 2 * HH) {
          #pragma unroll
          for (int r = 0; r < 8; ++r)
            sG[r * SG_STRIDE + g0 + ln15] += acc[j].f[r];
        } else {
          #pragma unroll
          for (int r = 0; r < 8; ++r)
            sNh[r * SN_STRIDE + (g0 - 2 * HH) + ln15] = acc[j].f[r];
        }
      }
    }
    __syncthreads();

    for (int i = 0; i < 16; ++i) {   // 8*512 elems
      int e = i * 256 + tid;
      int m = e >> 9, c = e & (HH - 1);
      float rr = sigmoidf_(sG[m * SG_STRIDE + c]);
      float zz = sigmoidf_(sG[m * SG_STRIDE + HH + c]);
      float nn = tanhf(sG[m * SG_STRIDE + 2 * HH + c] + rr * sNh[m * SN_STRIDE + c]);
      float hnew = (1.0f - zz) * nn + zz * sHf[m * SN_STRIDE + c];
      sHf[m * SN_STRIDE + c] = hnew;
      sHb[m * SH_STRIDE + c] = f2bf(hnew);
      doc_out[((size_t)(m * TDOC + t)) * DD + dir * HH + c] = hnew;
    }
    __syncthreads();
  }
}

// ---------------- scores + softmax + word broadcast ---------------------
__global__ void score_softmax_kernel(const float* __restrict__ doc_out,  // [8][128][1024]
                                     const float* __restrict__ Wf,       // [1024]
                                     const float* __restrict__ bfp,      // [1]
                                     float* __restrict__ out_total,      // [8*4*2048]
                                     float* __restrict__ out_dist)       // [8*128]
{
  __shared__ float red[128];
  const int b = blockIdx.x, t = threadIdx.x;  // 128 threads
  const float* row = doc_out + ((size_t)(b * TDOC + t)) * DD;
  float s = bfp[0];
  for (int d = 0; d < DD; ++d) s += row[d] * Wf[d];
  red[t] = s; __syncthreads();
  for (int off = 64; off > 0; off >>= 1) {
    if (t < off) red[t] = fmaxf(red[t], red[t + off]);
    __syncthreads();
  }
  float mx = red[0]; __syncthreads();
  float e = __expf(s - mx);
  red[t] = e; __syncthreads();
  for (int off = 64; off > 0; off >>= 1) {
    if (t < off) red[t] += red[t + off];
    __syncthreads();
  }
  float dist = e / red[0];
  out_dist[b * TDOC + t] = dist;
  int nsec = t >> 5, si = t & 31;
  float* base = out_total + ((size_t)(b * NSEC_ + nsec)) * SECSZ + si * LL;
  for (int w = 0; w < LL; ++w) base[w] = dist;
}

// ---------------- host-side orchestration -------------------------------
extern "C" void kernel_launch(void* const* d_in, const int* in_sizes, int n_in,
                              void* d_out, int out_size, void* d_ws, size_t ws_size,
                              hipStream_t stream) {
  const float* x      = (const float*)d_in[0];
  const float* Wih_sf = (const float*)d_in[2];
  const float* Whh_sf = (const float*)d_in[3];
  const float* bih_sf = (const float*)d_in[4];
  const float* bhh_sf = (const float*)d_in[5];
  const float* Wih_sb = (const float*)d_in[6];
  const float* Whh_sb = (const float*)d_in[7];
  const float* bih_sb = (const float*)d_in[8];
  const float* bhh_sb = (const float*)d_in[9];
  const float* Wih_df = (const float*)d_in[10];
  const float* Whh_df = (const float*)d_in[11];
  const float* bih_df = (const float*)d_in[12];
  const float* bhh_df = (const float*)d_in[13];
  const float* Wih_db = (const float*)d_in[14];
  const float* Whh_db = (const float*)d_in[15];
  const float* bih_db = (const float*)d_in[16];
  const float* bhh_db = (const float*)d_in[17];
  const float* Wf     = (const float*)d_in[18];
  const float* bfp    = (const float*)d_in[19];

  char* ws = (char*)d_ws;
  size_t off = 0;
  auto carve = [&](size_t bytes) -> char* {
    char* p = ws + off;
    off += (bytes + 255) & ~(size_t)255;
    return p;
  };
  const size_t NX = (size_t)NSENT * LL * DD;
  unsigned short* xbf     = (unsigned short*)carve(NX * 2);
  unsigned short* Wih_s   = (unsigned short*)carve((size_t)2 * D3 * DD * 2);
  unsigned short* Whh_s   = (unsigned short*)carve((size_t)2 * D3 * HH * 2);
  unsigned short* Wih_d   = (unsigned short*)carve((size_t)2 * D3 * DD * 2);
  unsigned short* Whh_d   = (unsigned short*)carve((size_t)2 * D3 * HH * 2);
  float* bias_s_comb      = (float*)carve((size_t)2 * D3 * 4);
  float* bias_s_hn        = (float*)carve((size_t)2 * HH * 4);
  float* bias_d_comb      = (float*)carve((size_t)2 * D3 * 4);
  float* bias_d_hn        = (float*)carve((size_t)2 * HH * 4);
  unsigned short* sent_h  = (unsigned short*)carve((size_t)NSENT * DD * 2);
  float* dg               = (float*)carve((size_t)2 * NSENT * D3 * 4);
  float* doc_out          = (float*)carve((size_t)NSENT * DD * 4);

  auto cvt = [&](const float* src, unsigned short* dst, size_t n) {
    int n8 = (int)(n / 8);
    cvt_bf16_kernel<<<(n8 + 255) / 256, 256, 0, stream>>>(src, dst, n8);
  };
  cvt(x, xbf, NX);
  cvt(Wih_sf, Wih_s,                       (size_t)D3 * DD);
  cvt(Wih_sb, Wih_s + (size_t)D3 * DD,     (size_t)D3 * DD);
  cvt(Whh_sf, Whh_s,                       (size_t)D3 * HH);
  cvt(Whh_sb, Whh_s + (size_t)D3 * HH,     (size_t)D3 * HH);
  cvt(Wih_df, Wih_d,                       (size_t)D3 * DD);
  cvt(Wih_db, Wih_d + (size_t)D3 * DD,     (size_t)D3 * DD);
  cvt(Whh_df, Whh_d,                       (size_t)D3 * HH);
  cvt(Whh_db, Whh_d + (size_t)D3 * HH,     (size_t)D3 * HH);

  bias_prep_kernel<<<6, 256, 0, stream>>>(bih_sf, bhh_sf, bias_s_comb,      bias_s_hn);
  bias_prep_kernel<<<6, 256, 0, stream>>>(bih_sb, bhh_sb, bias_s_comb + D3, bias_s_hn + HH);
  bias_prep_kernel<<<6, 256, 0, stream>>>(bih_df, bhh_df, bias_d_comb,      bias_d_hn);
  bias_prep_kernel<<<6, 256, 0, stream>>>(bih_db, bhh_db, bias_d_comb + D3, bias_d_hn + HH);

  sent_gru_kernel<<<128, 256, SENT_SMEM, stream>>>(xbf, Wih_s, Whh_s,
                                                   bias_s_comb, bias_s_hn, sent_h);

  doc_proj_kernel<<<dim3(96, 64, 2), 32, 0, stream>>>(sent_h, Wih_d, bias_d_comb, dg);

  doc_gru_kernel<<<2, 256, DOC_SMEM, stream>>>(dg, Whh_d, bias_d_hn, doc_out);

  float* out = (float*)d_out;
  score_softmax_kernel<<<BB, 128, 0, stream>>>(doc_out, Wf, bfp,
                                               out, out + (size_t)BB * NSEC_ * SECSZ);
}
